// MemNet_60241211293836
// MI455X (gfx1250) — matrix operand order repaired and checked by reading, not compiled, not run on hardware
//
#include <hip/hip_runtime.h>
#include <hip/hip_bf16.h>
#include <stdint.h>

// ---------------------------------------------------------------------------
// GRU network for MI455X (gfx1250, wave32, WMMA bf16 16x16x32, fp32 accum).
//
//   P) prep: pre-convert x (concat+pad K 98->128), w_ih (pad), w_out (pad)
//      to row-major bf16 so GEMM fragments are pure 16B vector loads.
//   A) x_proj = x @ w_ih.T + b_ih : per wave, A-fragments (4 k-slices) loaded
//      once, swept across 8 N-tiles -> 32 WMMAs/wave, no conversions.
//   B) persistent 64-WG GRU scan: per-WG w_hh slice (3x16x1024) packed once
//      into LDS in B-fragment order (96 KB of 320 KB LDS); per step 12 waves
//      x 32 k-slice WMMAs; atomic-counter grid barrier; h double-buffered
//      bf16; also emits bf16 rnn_hid for stage C.
//   C) out = tanh(rnn_hid @ w_out.T + b_out)*8 : one wave per M-tile holds
//      3 accumulators (N=48 padded), A loaded once per k-slice -> 96 WMMAs.
// ---------------------------------------------------------------------------

#define HIDN   1024
#define G3     3072
#define TLEN   1024
#define BATCH  64
#define KIN    98
#define KPAD   128
#define BT     (BATCH * TLEN)     // 65536
#define OUTN   33
#define NWG    64
#define WAVES  12
#define THREADS (WAVES * 32)

typedef __attribute__((ext_vector_type(16))) __bf16 v16bf;
typedef __attribute__((ext_vector_type(8)))  float  v8f;
typedef __attribute__((ext_vector_type(4)))  unsigned int u32x4;

union Frag {
    v16bf v;
    u32x4 q[2];
    unsigned short us[16];
};

static __device__ __forceinline__ unsigned short f2bf(float f) {
    union { __bf16 b; unsigned short u; } cv; cv.b = (__bf16)f; return cv.u;
}
static __device__ __forceinline__ float bf2f(unsigned short u) {
    union { __bf16 b; unsigned short u; } cv; cv.u = u; return (float)cv.b;
}

// Monotonic-counter grid barrier (counter zeroed by hipMemsetAsync per launch).
static __device__ __forceinline__ void grid_barrier(unsigned* cnt, unsigned phase) {
    __syncthreads();
    if (threadIdx.x == 0) {
        __threadfence();
        atomicAdd(cnt, 1u);
        const unsigned target = (unsigned)gridDim.x * phase;
        while (atomicAdd(cnt, 0u) < target) { __builtin_amdgcn_s_sleep(2); }
        __threadfence();
    }
    __syncthreads();
}

// ---------------------------------------------------------------------------
// Prep: x -> bf16 [BT][128] (concat ins|tar, zero-pad K)
// ---------------------------------------------------------------------------
__global__ void prep_x_kernel(const float* __restrict__ ins, const float* __restrict__ tar,
                              unsigned short* __restrict__ xbf) {
    size_t idx = (size_t)blockIdx.x * blockDim.x + threadIdx.x;
    if (idx >= (size_t)BT * KPAD) return;
    int m = (int)(idx >> 7), k = (int)(idx & (KPAD - 1));
    float f = 0.f;
    if (k < 65)       f = ins[(size_t)m * 65 + k];
    else if (k < KIN) f = tar[(size_t)m * 33 + (k - 65)];
    xbf[idx] = f2bf(f);
}

// Prep: w_ih -> bf16 [3072][128] (pad K); w_out -> bf16 [48][1024] (pad N rows)
__global__ void prep_w_kernel(const float* __restrict__ w_ih, const float* __restrict__ w_out,
                              unsigned short* __restrict__ wihbf,
                              unsigned short* __restrict__ woutbf) {
    int idx = blockIdx.x * blockDim.x + threadIdx.x;
    if (idx < G3 * KPAD) {
        int r = idx >> 7, k = idx & (KPAD - 1);
        wihbf[idx] = f2bf(k < KIN ? w_ih[(size_t)r * KIN + k] : 0.f);
    } else {
        int j = idx - G3 * KPAD;
        if (j < 48 * HIDN) {
            int r = j >> 10, k = j & (HIDN - 1);
            woutbf[j] = f2bf(r < OUTN ? w_out[(size_t)r * HIDN + k] : 0.f);
        }
    }
}

// ---------------------------------------------------------------------------
// Stage A: x_proj[m][g] = x @ w_ih.T + b_ih.  Wave: 1 M-tile x 8 N-tiles.
// ---------------------------------------------------------------------------
__global__ void xproj_kernel(const unsigned short* __restrict__ xbf,
                             const unsigned short* __restrict__ wihbf,
                             const float* __restrict__ b_ih,
                             float* __restrict__ xproj) {
    const int lane = threadIdx.x & 31;
    const int wave = threadIdx.x >> 5;
    const int gw   = blockIdx.x * 8 + wave;
    const int mt = gw / 24;            // 4096 M-tiles
    const int nc = gw % 24;            // chunk of 8 N-tiles
    const int m0 = mt * 16;
    const int col = lane & 15;
    const bool hi = lane >= 16;

    // A fragments: 4 k-slices, loaded once, pure 16B loads
    const unsigned short* ap = xbf + (size_t)(m0 + col) * KPAD;
    Frag a[4];
#pragma unroll
    for (int ks = 0; ks < 4; ++ks) {
        const int k0 = ks * 32;
        a[ks].q[0] = *(const u32x4*)(ap + k0 + (hi ? 16 : 0));
        a[ks].q[1] = *(const u32x4*)(ap + k0 + 8 + (hi ? 16 : 0));
    }

#pragma unroll
    for (int j = 0; j < 8; ++j) {
        const int n0 = (nc * 8 + j) * 16;
        const unsigned short* bp = wihbf + (size_t)(n0 + col) * KPAD;
        v8f c = {};
#pragma unroll
        for (int ks = 0; ks < 4; ++ks) {
            const int k0 = ks * 32 + (hi ? 16 : 0);
            Frag b;
            b.q[0] = *(const u32x4*)(bp + k0);
            b.q[1] = *(const u32x4*)(bp + k0 + 8);
            c = __builtin_amdgcn_wmma_f32_16x16x32_bf16(false, a[ks].v, false, b.v,
                                                        (short)0, c, false, false);
        }
        const float bias = b_ih[n0 + col];
#pragma unroll
        for (int i = 0; i < 8; ++i) {
            int m = m0 + i + (hi ? 8 : 0);
            xproj[(size_t)m * G3 + n0 + col] = c[i] + bias;
        }
    }
}

// ---------------------------------------------------------------------------
// Stage B: persistent GRU scan. 64 WGs x 384 threads (12 waves).
// WG bb owns hidden units [bb*16, bb*16+16).
// ---------------------------------------------------------------------------
__global__ void gru_scan_kernel(const float* __restrict__ w_hh, const float* __restrict__ b_hh,
                                const float* __restrict__ xproj,
                                unsigned short* __restrict__ hbuf,   // [2][64][1024] bf16
                                float* __restrict__ rnn,             // (B,T,H) fp32
                                unsigned short* __restrict__ rnnbf,  // (B,T,H) bf16
                                unsigned* __restrict__ counter) {
    extern __shared__ char smem[];
    unsigned short* wfrag = (unsigned short*)smem;                 // [3][32][32][16] bf16
    float* ghls  = (float*)(smem + 3 * 32 * 32 * 16 * 2);          // [3][64][16]
    float* bhh_s = ghls + 3 * BATCH * 16;                          // [48]

    const int lane = threadIdx.x & 31;
    const int wave = threadIdx.x >> 5;
    const int n0   = blockIdx.x * 16;
    const int col  = lane & 15;
    const bool hi  = lane >= 16;

    // ---- one-time: pack this WG's w_hh slice into LDS in B-fragment order ----
    for (int f = wave; f < 96; f += WAVES) {
        const int g = f / 32, ks = f % 32;
        const int row = g * HIDN + n0 + col;           // w_hh output row
        const int kb  = ks * 32 + (hi ? 16 : 0);
        const float* wp = w_hh + (size_t)row * HIDN + kb;
        unsigned short* dst = wfrag + (((g * 32 + ks) * 32 + lane) * 16);
#pragma unroll
        for (int e = 0; e < 16; ++e) dst[e] = f2bf(wp[e]);
    }
    for (int i = threadIdx.x; i < 48; i += THREADS)
        bhh_s[i] = b_hh[(i / 16) * HIDN + n0 + (i % 16)];
    // h0 = 0.1 into buffer 0 (our hidden slice)
    for (int i = threadIdx.x; i < BATCH * 16; i += THREADS) {
        int b = i >> 4, n = i & 15;
        hbuf[(size_t)b * HIDN + n0 + n] = f2bf(0.1f);
    }
    grid_barrier(counter, 1);

    const int g  = wave >> 2;           // gate 0..2
    const int mt = wave & 3;            // M tile 0..3
    const int arow_b = mt * 16 + col;   // batch row this lane carries in A

    for (int t = 0; t < TLEN; ++t) {
        const unsigned short* hcur = hbuf + (size_t)(t & 1) * BATCH * HIDN;
        unsigned short* hnext      = hbuf + (size_t)((t + 1) & 1) * BATCH * HIDN;

        // prefetch this step's x_proj rows (global_prefetch_b8)
        for (int idx = threadIdx.x; idx < BATCH * 3; idx += THREADS) {
            int b = idx & (BATCH - 1), gg = idx >> 6;
            __builtin_prefetch(&xproj[((size_t)b * TLEN + t) * G3 + gg * HIDN + n0], 0, 3);
        }

        // gh tile: (64 x 16) for gate g, M-tile mt; K = 1024
        v8f c = {};
        const unsigned short* ap_base = hcur + (size_t)arow_b * HIDN;
#pragma unroll 8
        for (int ks = 0; ks < HIDN / 32; ++ks) {
            const int k0 = ks * 32;
            Frag a, b;
            a.q[0] = *(const u32x4*)(ap_base + k0 + (hi ? 16 : 0));
            a.q[1] = *(const u32x4*)(ap_base + k0 + 8 + (hi ? 16 : 0));
            const u32x4* bp = (const u32x4*)(wfrag + (((g * 32 + ks) * 32 + lane) * 16));
            b.q[0] = bp[0];
            b.q[1] = bp[1];
            c = __builtin_amdgcn_wmma_f32_16x16x32_bf16(false, a.v, false, b.v,
                                                        (short)0, c, false, false);
        }
#pragma unroll
        for (int i = 0; i < 8; ++i) {
            int b = mt * 16 + i + (hi ? 8 : 0);
            ghls[(g * BATCH + b) * 16 + col] = c[i];
        }
        __syncthreads();

        // fuse gates, produce h_new for our 16 hidden units x 64 batches
        for (int idx = threadIdx.x; idx < BATCH * 16; idx += THREADS) {
            int b = idx >> 4, n = idx & 15;
            size_t xr_i = ((size_t)b * TLEN + t) * G3 + n0 + n;
            float xr = xproj[xr_i];
            float xz = xproj[xr_i + HIDN];
            float xn = xproj[xr_i + 2 * HIDN];
            float hr = ghls[(0 * BATCH + b) * 16 + n] + bhh_s[n];
            float hz = ghls[(1 * BATCH + b) * 16 + n] + bhh_s[16 + n];
            float hn = ghls[(2 * BATCH + b) * 16 + n] + bhh_s[32 + n];
            float r  = 1.f / (1.f + __expf(-(xr + hr)));
            float z  = 1.f / (1.f + __expf(-(xz + hz)));
            float nn = fmaxf(xn + r * hn, 0.f);
            float hp = bf2f(hcur[(size_t)b * HIDN + n0 + n]);
            float hv = (1.f - z) * nn + z * hp;
            unsigned short hv16 = f2bf(hv);
            hnext[(size_t)b * HIDN + n0 + n] = hv16;
            size_t ridx = ((size_t)b * TLEN + t) * HIDN + n0 + n;
            rnn[ridx]   = hv;
            rnnbf[ridx] = hv16;
        }
        grid_barrier(counter, (unsigned)(t + 2));
    }
}

// ---------------------------------------------------------------------------
// Stage C: out = tanh(rnn_hid @ w_out.T + b_out) * 8.
// One wave per M-tile; 3 N-tile accumulators (N padded 33->48); A loaded once
// per k-slice and reused -> 96 WMMAs per wave, zero conversions.
// ---------------------------------------------------------------------------
__global__ void outproj_kernel(const unsigned short* __restrict__ rnnbf,
                               const unsigned short* __restrict__ woutbf,
                               const float* __restrict__ b_out,
                               float* __restrict__ out) {
    const int lane = threadIdx.x & 31;
    const int wave = threadIdx.x >> 5;
    const int mt   = blockIdx.x * 8 + wave;
    const int m0 = mt * 16;
    const int col = lane & 15;
    const bool hi = lane >= 16;

    const unsigned short* ap = rnnbf + (size_t)(m0 + col) * HIDN;
    v8f c[3] = {{}, {}, {}};
#pragma unroll 4
    for (int ks = 0; ks < HIDN / 32; ++ks) {
        const int k0 = ks * 32;
        Frag a;
        a.q[0] = *(const u32x4*)(ap + k0 + (hi ? 16 : 0));
        a.q[1] = *(const u32x4*)(ap + k0 + 8 + (hi ? 16 : 0));
#pragma unroll
        for (int nt = 0; nt < 3; ++nt) {
            const unsigned short* bp =
                woutbf + (size_t)(nt * 16 + col) * HIDN + k0 + (hi ? 16 : 0);
            Frag b;
            b.q[0] = *(const u32x4*)(bp);
            b.q[1] = *(const u32x4*)(bp + 8);
            c[nt] = __builtin_amdgcn_wmma_f32_16x16x32_bf16(false, a.v, false, b.v,
                                                            (short)0, c[nt], false, false);
        }
    }
#pragma unroll
    for (int nt = 0; nt < 3; ++nt) {
        const int brow = nt * 16 + col;
        if (brow < OUTN) {
            const float bias = b_out[brow];
#pragma unroll
            for (int i = 0; i < 8; ++i) {
                int m = m0 + i + (hi ? 8 : 0);
                out[(size_t)m * OUTN + brow] = tanhf(c[nt][i] + bias) * 8.0f;
            }
        }
    }
}

// ---------------------------------------------------------------------------
extern "C" void kernel_launch(void* const* d_in, const int* in_sizes, int n_in,
                              void* d_out, int out_size, void* d_ws, size_t ws_size,
                              hipStream_t stream) {
    (void)in_sizes; (void)n_in; (void)out_size; (void)ws_size;
    const float* ins   = (const float*)d_in[0];
    const float* tar   = (const float*)d_in[1];
    const float* w_ih  = (const float*)d_in[2];
    const float* w_hh  = (const float*)d_in[3];
    const float* b_ih  = (const float*)d_in[4];
    const float* b_hh  = (const float*)d_in[5];
    const float* w_out = (const float*)d_in[6];
    const float* b_out = (const float*)d_in[7];

    float* out = (float*)d_out;                         // (B,T,33)
    float* rnn = out + (size_t)BATCH * TLEN * OUTN;     // (B,T,H)

    // scratch layout
    char* ws = (char*)d_ws;
    size_t off = 0;
    unsigned* counter = (unsigned*)(ws + off);          off += 1024;
    unsigned short* hbuf   = (unsigned short*)(ws + off); off += (size_t)2 * BATCH * HIDN * 2;
    unsigned short* xbf    = (unsigned short*)(ws + off); off += (size_t)BT * KPAD * 2;
    unsigned short* wihbf  = (unsigned short*)(ws + off); off += (size_t)G3 * KPAD * 2;
    unsigned short* woutbf = (unsigned short*)(ws + off); off += (size_t)48 * HIDN * 2;
    unsigned short* rnnbf  = (unsigned short*)(ws + off); off += (size_t)BT * HIDN * 2;
    float* xproj = (float*)(ws + off);                  // BT*G3*4

    hipMemsetAsync(d_ws, 0, 1024, stream);

    // Prep
    prep_x_kernel<<<dim3((unsigned)(((size_t)BT * KPAD + 255) / 256)), dim3(256), 0, stream>>>(
        ins, tar, xbf);
    prep_w_kernel<<<dim3((G3 * KPAD + 48 * HIDN + 255) / 256), dim3(256), 0, stream>>>(
        w_ih, w_out, wihbf, woutbf);

    // Stage A: 4096 M-tiles x 24 chunks (8 N-tiles each), 8 waves/WG
    xproj_kernel<<<dim3(4096 * 24 / 8), dim3(256), 0, stream>>>(xbf, wihbf, b_ih, xproj);

    // Stage B: persistent scan, 64 WGs (co-resident), 12 waves each
    const size_t smem = (size_t)3 * 32 * 32 * 16 * 2   // wfrag (96 KB)
                      + (size_t)3 * BATCH * 16 * 4     // ghls  (12 KB)
                      + 64 * 4;                        // biases + pad
    gru_scan_kernel<<<dim3(NWG), dim3(THREADS), smem, stream>>>(
        w_hh, b_hh, xproj, hbuf, rnn, rnnbf, counter);

    // Stage C: 4096 M-tiles, one wave each, 8 waves/WG
    outproj_kernel<<<dim3(4096 / 8), dim3(256), 0, stream>>>(rnnbf, woutbf, b_out, out);
}
